// GCNStandardSupervised_37821482009256
// MI455X (gfx1250) — matrix operand order repaired and checked by reading, not compiled, not run on hardware
//
#include <hip/hip_runtime.h>
#include <hip/hip_bf16.h>

typedef __attribute__((ext_vector_type(16))) _Float16 v16h;
typedef __attribute__((ext_vector_type(8)))  float    v8f;

#define FDIM 128
#define EPSV 1e-5f

// ---------------- utility ----------------
__global__ void fill_kernel(float* __restrict__ p, float v, size_t n) {
    size_t i = (size_t)blockIdx.x * blockDim.x + threadIdx.x;
    if (i < n) p[i] = v;
}

// ---------------- degree / normalization ----------------
__global__ void deg_edges_kernel(const int* __restrict__ dst, float* __restrict__ deg, int E) {
    int e = blockIdx.x * blockDim.x + threadIdx.x;
    if (e < E) unsafeAtomicAdd(&deg[dst[e]], 1.0f);
}

__global__ void deg_finalize_kernel(const float* __restrict__ deg,
                                    float* __restrict__ dinv, float* __restrict__ snorm, int N) {
    int i = blockIdx.x * blockDim.x + threadIdx.x;
    if (i < N) {
        float d = deg[i];          // already includes +1 self loop (init fill = 1.0)
        dinv[i]  = rsqrtf(d);
        snorm[i] = 1.0f / d;
    }
}

// ---------------- W (f32, [K,C]) -> Wt (f16, [C,K]) ----------------
__global__ void convert_wt_kernel(const float* __restrict__ W, _Float16* __restrict__ Wt) {
    int i = blockIdx.x * blockDim.x + threadIdx.x;   // 128*128 threads
    int k = i >> 7, c = i & 127;
    Wt[c * FDIM + k] = (_Float16)W[k * FDIM + c];
}

// ---------------- GEMM: Out[M,128] = A[M,128] @ W  via v_wmma_f32_16x16x32_f16 ----
// grid.x = M/16 (M multiple of 16), block = 256 threads = 8 waves.
// Wave w computes the 16x16 tile at columns [16w, 16w+16).
__global__ void __launch_bounds__(256)
gemm_wmma_kernel(const float* __restrict__ A, const _Float16* __restrict__ Wt,
                 float* __restrict__ Out, int M) {
    const int lane = threadIdx.x & 31;
    const int wave = threadIdx.x >> 5;
    const int m0   = blockIdx.x * 16;
    const int l15  = lane & 15;
    const int hi   = lane >> 4;             // 0 for lanes 0-15, 1 for lanes 16-31
    const int row  = m0 + l15;              // A-fragment row for this lane
    const int col  = wave * 16 + l15;       // B/C-fragment column for this lane
    const int koffA   = hi * 16;            // A: hi half-wave holds K+16..31
    const int khalfB  = hi * 16;            // B: hi half-wave holds K+16..31

    v8f acc = {};
    #pragma unroll
    for (int kk = 0; kk < FDIM; kk += 32) {
        // A fragment: per-lane elements 0..15 = K(kk+koffA+0..15)
        const float* arow = A + (size_t)row * FDIM + kk + koffA;
        float4 p0 = *(const float4*)(arow + 0);
        float4 p1 = *(const float4*)(arow + 4);
        float4 p2 = *(const float4*)(arow + 8);
        float4 p3 = *(const float4*)(arow + 12);
        v16h a;
        a[0] = (_Float16)p0.x;  a[1] = (_Float16)p0.y;  a[2]  = (_Float16)p0.z;  a[3]  = (_Float16)p0.w;
        a[4] = (_Float16)p1.x;  a[5] = (_Float16)p1.y;  a[6]  = (_Float16)p1.z;  a[7]  = (_Float16)p1.w;
        a[8] = (_Float16)p2.x;  a[9] = (_Float16)p2.y;  a[10] = (_Float16)p2.z;  a[11] = (_Float16)p2.w;
        a[12] = (_Float16)p3.x; a[13] = (_Float16)p3.y; a[14] = (_Float16)p3.z;  a[15] = (_Float16)p3.w;
        // B fragment: 16 contiguous K values of column `col` (Wt is [C,K])
        v16h b = *(const v16h*)(Wt + (size_t)col * FDIM + kk + khalfB);
        acc = __builtin_amdgcn_wmma_f32_16x16x32_f16(
            /*neg_a=*/false, a, /*neg_b=*/false, b,
            /*c_mod=*/(short)0, acc, /*reuse_a=*/false, /*reuse_b=*/false);
    }
    // C/D layout: VGPR r -> M = m0 + hi*8 + r, N = col
    #pragma unroll
    for (int r = 0; r < 8; ++r)
        Out[(size_t)(m0 + hi * 8 + r) * FDIM + col] = acc[r];
}

// ---------------- aggregation: out = h*snorm (self) + scatter(h[src]*enorm) ----
__global__ void agg_init_kernel(const float* __restrict__ h, const float* __restrict__ snorm,
                                float* __restrict__ out, size_t n) {
    size_t i = (size_t)blockIdx.x * blockDim.x + threadIdx.x;
    if (i < n) out[i] = h[i] * snorm[i >> 7];
}

__global__ void agg_edges_kernel(const float* __restrict__ h,
                                 const int* __restrict__ src, const int* __restrict__ dst,
                                 const float* __restrict__ dinv,
                                 float* __restrict__ out, int E) {
    int e = blockIdx.x * (blockDim.x >> 5) + (threadIdx.x >> 5);
    if (e >= E) return;
    int lane = threadIdx.x & 31;
    int s = src[e], d = dst[e];
    float w = dinv[s] * dinv[d];
    const float* hs = h + (size_t)s * FDIM;
    float* od = out + (size_t)d * FDIM;
    #pragma unroll
    for (int j = 0; j < FDIM; j += 32)
        unsafeAtomicAdd(&od[j + lane], hs[j + lane] * w);
}

// ---------------- BatchNorm (training stats) + ReLU ----------------
#define BN_ROWS_PER_BLOCK 256
__global__ void bn_reduce_kernel(const float* __restrict__ h,
                                 float* __restrict__ sums, float* __restrict__ sumsq, int rows) {
    int f  = threadIdx.x & 127;
    int rr = threadIdx.x >> 7;                 // 0..1
    int r0 = blockIdx.x * BN_ROWS_PER_BLOCK;
    int r1 = r0 + BN_ROWS_PER_BLOCK; if (r1 > rows) r1 = rows;
    float s = 0.f, q = 0.f;
    for (int r = r0 + rr; r < r1; r += 2) {
        float v = h[(size_t)r * FDIM + f];
        s += v; q += v * v;
    }
    unsafeAtomicAdd(&sums[f], s);
    unsafeAtomicAdd(&sumsq[f], q);
}

__global__ void bn_apply_relu_kernel(float* __restrict__ h,
                                     const float* __restrict__ sums, const float* __restrict__ sumsq,
                                     const float* __restrict__ gam, const float* __restrict__ bet,
                                     float invR, size_t n) {
    size_t i = (size_t)blockIdx.x * blockDim.x + threadIdx.x;
    if (i >= n) return;
    int f = i & 127;
    float mean = sums[f] * invR;
    float var  = sumsq[f] * invR - mean * mean;
    float sc   = gam[f] * rsqrtf(var + EPSV);
    float v    = (h[i] - mean) * sc + bet[f];
    h[i] = fmaxf(v, 0.0f);
}

// ---------------- global mean pool ----------------
__global__ void pool_count_kernel(const int* __restrict__ batch, float* __restrict__ counts, int N) {
    int i = blockIdx.x * blockDim.x + threadIdx.x;
    if (i < N) unsafeAtomicAdd(&counts[batch[i]], 1.0f);
}

__global__ void pool_add_kernel(const float* __restrict__ h, const int* __restrict__ batch,
                                float* __restrict__ pooled, size_t n) {
    size_t i = (size_t)blockIdx.x * blockDim.x + threadIdx.x;
    if (i >= n) return;
    int r = i >> 7, f = i & 127;
    unsafeAtomicAdd(&pooled[(size_t)batch[r] * FDIM + f], h[i]);
}

__global__ void pool_div_kernel(float* __restrict__ pooled, const float* __restrict__ counts, int n) {
    int i = blockIdx.x * blockDim.x + threadIdx.x;
    if (i < n) pooled[i] /= fmaxf(counts[i >> 7], 1.0f);
}

// ---------------- head BN (+ReLU) over few rows, 1 block / 128 threads ----------
__global__ void bn_small_kernel(float* __restrict__ z, const float* __restrict__ gam,
                                const float* __restrict__ bet, int rows) {
    int f = threadIdx.x;                      // 128
    float s = 0.f, q = 0.f;
    for (int r = 0; r < rows; ++r) { float v = z[r * FDIM + f]; s += v; q += v * v; }
    float invR = 1.0f / rows;
    float mean = s * invR, var = q * invR - mean * mean;
    float sc = gam[f] * rsqrtf(var + EPSV);
    for (int r = 0; r < rows; ++r) {
        float v = (z[r * FDIM + f] - mean) * sc + bet[f];
        z[r * FDIM + f] = fmaxf(v, 0.0f);
    }
}

// ---------------- final Linear(128,1) + BN(1): 1 block, G threads ----------------
__global__ void head_final_kernel(const float* __restrict__ z, const float* __restrict__ w1,
                                  const float* __restrict__ gam, const float* __restrict__ bet,
                                  float* __restrict__ out, int rows) {
    __shared__ float t[128];
    int r = threadIdx.x;
    float acc = 0.f;
    for (int k = 0; k < FDIM; ++k) acc += z[r * FDIM + k] * w1[k];
    t[r] = acc;
    __syncthreads();
    float s = 0.f, q = 0.f;
    for (int i = 0; i < rows; ++i) { s += t[i]; q += t[i] * t[i]; }
    float invR = 1.0f / rows;
    float mean = s * invR, var = q * invR - mean * mean;
    out[r] = (acc - mean) * gam[0] * rsqrtf(var + EPSV) + bet[0];
}

// =====================================================================
extern "C" void kernel_launch(void* const* d_in, const int* in_sizes, int n_in,
                              void* d_out, int out_size, void* d_ws, size_t ws_size,
                              hipStream_t stream) {
    const float* x     = (const float*)d_in[0];
    const int*   eidx  = (const int*)d_in[1];
    const int*   batch = (const int*)d_in[2];
    // params: conv{i}_W, conv{i}_b, bn{i}_g, bn{i}_b for i=0..2, then head
    const float* convW[3]; const float* bnG[3]; const float* bnB[3];
    for (int i = 0; i < 3; ++i) {
        convW[i] = (const float*)d_in[3 + 4 * i + 0];
        bnG[i]   = (const float*)d_in[3 + 4 * i + 2];
        bnB[i]   = (const float*)d_in[3 + 4 * i + 3];
    }
    const float* lin0W = (const float*)d_in[15];
    const float* lbn0g = (const float*)d_in[17];
    const float* lbn0b = (const float*)d_in[18];
    const float* lin1W = (const float*)d_in[19];
    const float* lbn1g = (const float*)d_in[21];
    const float* lbn1b = (const float*)d_in[22];

    const int N = in_sizes[2];           // 50000 (multiple of 16)
    const int E = in_sizes[1] / 2;       // 1600000
    const int G = out_size;              // 64
    const int* src = eidx;
    const int* dst = eidx + E;
    const size_t NF = (size_t)N * FDIM;
    const size_t Npad = (size_t)((N + 63) & ~63);

    // workspace carve-up (floats; base is 256B aligned from hipMalloc)
    float* hA     = (float*)d_ws;        // gemm output        [N,128]
    float* hB     = hA + NF;             // aggregated/activ.  [N,128]
    float* deg    = hB + NF;             // [N]
    float* dinv   = deg + Npad;          // [N]
    float* snorm  = dinv + Npad;         // [N]
    float* sums   = snorm + Npad;        // [128]
    float* sumsq  = sums + 128;          // [128]
    float* pooled = sumsq + 128;         // [G,128]
    float* counts = pooled + (size_t)G * FDIM;   // [G]
    float* zbuf   = counts + 64;         // [G,128]
    _Float16* Wth = (_Float16*)(zbuf + (size_t)G * FDIM);  // [128,128] f16 (transposed)

    const int T = 256;
    const float invN = 1.0f / (float)N;

    // --- degrees: deg = 1 (self loop) + in-degree; dinv = rsqrt(deg); snorm = 1/deg
    fill_kernel<<<(N + T - 1) / T, T, 0, stream>>>(deg, 1.0f, (size_t)N);
    deg_edges_kernel<<<(E + T - 1) / T, T, 0, stream>>>(dst, deg, E);
    deg_finalize_kernel<<<(N + T - 1) / T, T, 0, stream>>>(deg, dinv, snorm, N);

    // --- 3 GCN layers (conv bias skipped: cancelled exactly by following BN)
    const float* hin = x;
    for (int l = 0; l < 3; ++l) {
        convert_wt_kernel<<<(FDIM * FDIM) / T, T, 0, stream>>>(convW[l], Wth);
        gemm_wmma_kernel<<<N / 16, T, 0, stream>>>(hin, Wth, hA, N);
        agg_init_kernel<<<(int)((NF + T - 1) / T), T, 0, stream>>>(hA, snorm, hB, NF);
        agg_edges_kernel<<<(E + 7) / 8, T, 0, stream>>>(hA, src, dst, dinv, hB, E);
        fill_kernel<<<1, T, 0, stream>>>(sums, 0.0f, 256);   // sums+sumsq contiguous
        bn_reduce_kernel<<<(N + BN_ROWS_PER_BLOCK - 1) / BN_ROWS_PER_BLOCK, T, 0, stream>>>(hB, sums, sumsq, N);
        bn_apply_relu_kernel<<<(int)((NF + T - 1) / T), T, 0, stream>>>(hB, sums, sumsq, bnG[l], bnB[l], invN, NF);
        hin = hB;
    }

    // --- global mean pool
    fill_kernel<<<((G * FDIM + 64) + T - 1) / T, T, 0, stream>>>(pooled, 0.0f, (size_t)G * FDIM + 64);
    pool_count_kernel<<<(N + T - 1) / T, T, 0, stream>>>(batch, counts, N);
    pool_add_kernel<<<(int)((NF + T - 1) / T), T, 0, stream>>>(hB, batch, pooled, NF);
    pool_div_kernel<<<(G * FDIM + T - 1) / T, T, 0, stream>>>(pooled, counts, G * FDIM);

    // --- head: Linear(128,128) (bias cancels in BN) -> BN -> ReLU
    convert_wt_kernel<<<(FDIM * FDIM) / T, T, 0, stream>>>(lin0W, Wth);
    gemm_wmma_kernel<<<G / 16, T, 0, stream>>>(pooled, Wth, zbuf, G);
    bn_small_kernel<<<1, FDIM, 0, stream>>>(zbuf, lbn0g, lbn0b, G);

    // --- Linear(128,1) (bias cancels) -> BN(1)
    head_final_kernel<<<1, G, 0, stream>>>(zbuf, lin1W, lbn1g, lbn1b, (float*)d_out, G);
}
